// MultiHeadAttention_292057776516
// MI455X (gfx1250) — matrix operand order repaired and checked by reading, not compiled, not run on hardware
//
#include <hip/hip_runtime.h>
#include <hip/hip_bf16.h>

typedef __attribute__((ext_vector_type(16))) _Float16 v16h;
typedef __attribute__((ext_vector_type(8)))  _Float16 v8h;
typedef __attribute__((ext_vector_type(8)))  float    v8f;

#define D_MODEL 512
#define SEQ 2048
#define NBATCH 4
#define NHEADS 8
#define MTOT (NBATCH * SEQ)          // 8192 rows
#define NEG_INF (-3.0e38f)

union AFrag { v16h v; v8h h[2]; };
union H8   { v8h v; _Float16 e[8]; };

__device__ __forceinline__ v8f wmma_f16(const AFrag& a, const AFrag& b, v8f c) {
  return __builtin_amdgcn_wmma_f32_16x16x32_f16(false, a.v, false, b.v, (short)0, c, false, false);
}

__device__ __forceinline__ float redmax16(float v) {
  #pragma unroll
  for (int off = 8; off; off >>= 1) v = fmaxf(v, __shfl_xor(v, off, 32));
  return v;
}
__device__ __forceinline__ float redsum16(float v) {
  #pragma unroll
  for (int off = 8; off; off >>= 1) v += __shfl_xor(v, off, 32);
  return v;
}

// LDS byte offset of a __shared__ object (generic -> addrspace(3) cast)
__device__ __forceinline__ unsigned lds_off(const void* p) {
  return (unsigned)(unsigned long long)(const __attribute__((address_space(3))) void*)p;
}

// async global -> LDS 16B copy (ASYNCcnt-tracked, bypasses VGPRs)
__device__ __forceinline__ void async_copy_b128(unsigned lds_dst, const void* gsrc) {
  asm volatile("global_load_async_to_lds_b128 %0, %1, off"
               :: "v"(lds_dst), "v"(gsrc) : "memory");
}
__device__ __forceinline__ void wait_async0() {
  asm volatile("s_wait_asynccnt 0x0" ::: "memory");
}

// ---------------- weight transpose + f16 convert: W[In][Out] -> Wt[Out][In] ----------------
__global__ void cvt_w_kernel(const float* __restrict__ W, _Float16* __restrict__ Wt,
                             int In, int Out) {
  int i = blockIdx.x * blockDim.x + threadIdx.x;
  if (i >= In * Out) return;
  int r = i / Out, c = i - r * Out;
  Wt[(size_t)c * In + r] = (_Float16)W[i];
}

// ---------------- LayerNorm: one wave per 512-wide row, output f16 ----------------
__launch_bounds__(128)
__global__ void ln_kernel(const float* __restrict__ x, const float* __restrict__ gamma,
                          const float* __restrict__ beta, _Float16* __restrict__ xn) {
  int lane = threadIdx.x & 31;
  int row  = blockIdx.x * 4 + (threadIdx.x >> 5);
  const float* xr = x + (size_t)row * D_MODEL;
  float4 v[4];
  float s = 0.f, ss = 0.f;
  #pragma unroll
  for (int i = 0; i < 4; ++i) {
    v[i] = ((const float4*)xr)[lane * 4 + i];
    s  += v[i].x + v[i].y + v[i].z + v[i].w;
    ss += v[i].x*v[i].x + v[i].y*v[i].y + v[i].z*v[i].z + v[i].w*v[i].w;
  }
  #pragma unroll
  for (int off = 16; off; off >>= 1) { s += __shfl_xor(s, off, 32); ss += __shfl_xor(ss, off, 32); }
  float mu = s * (1.0f / D_MODEL);
  float var = ss * (1.0f / D_MODEL) - mu * mu;
  float rs = rsqrtf(var + 1e-5f);
  _Float16* outp = xn + (size_t)row * D_MODEL + lane * 16;
  H8 o[2];
  #pragma unroll
  for (int i = 0; i < 4; ++i) {
    float4 g = ((const float4*)gamma)[lane * 4 + i];
    float4 b = ((const float4*)beta )[lane * 4 + i];
    float r0 = (v[i].x - mu) * rs * g.x + b.x;
    float r1 = (v[i].y - mu) * rs * g.y + b.y;
    float r2 = (v[i].z - mu) * rs * g.z + b.z;
    float r3 = (v[i].w - mu) * rs * g.w + b.w;
    int u = i >> 1;
    int base = (i & 1) * 4;
    o[u].e[base+0] = (_Float16)r0; o[u].e[base+1] = (_Float16)r1;
    o[u].e[base+2] = (_Float16)r2; o[u].e[base+3] = (_Float16)r3;
  }
  *(v8h*)(outp + 0) = o[0].v;
  *(v8h*)(outp + 8) = o[1].v;
}

// ---------------- QKV projection GEMM: Xn[8192][512]f16 @ Wt[1536][512]f16 ----------------
// 8 waves, block tile 128(M) x 64(N), K in 64-chunks, async-to-LDS double buffered.
__launch_bounds__(256)
__global__ void gemm_qkv_kernel(const _Float16* __restrict__ Xn, const _Float16* __restrict__ Wt,
                                const float* __restrict__ bq, const float* __restrict__ bkv,
                                _Float16* __restrict__ Qb, _Float16* __restrict__ Kb,
                                _Float16* __restrict__ Vt) {
  __shared__ __align__(16) _Float16 As[2][128][64];
  __shared__ __align__(16) _Float16 Bs[2][64][64];
  const int m0 = blockIdx.x * 128;
  const int n0 = blockIdx.y * 64;
  const int tid = threadIdx.x;
  const int lane = tid & 31, w = tid >> 5;
  const int half = lane >> 4, lrow = lane & 15;

  // prologue: stage K-chunk 0 into buffer 0
  #pragma unroll
  for (int i = 0; i < 4; ++i) {
    int lin = tid + i * 256, row = lin >> 3, c8 = lin & 7;
    async_copy_b128(lds_off(&As[0][row][c8 * 8]),
                    Xn + (size_t)(m0 + row) * D_MODEL + c8 * 8);
  }
  #pragma unroll
  for (int i = 0; i < 2; ++i) {
    int lin = tid + i * 256, row = lin >> 3, c8 = lin & 7;
    async_copy_b128(lds_off(&Bs[0][row][c8 * 8]),
                    Wt + (size_t)(n0 + row) * D_MODEL + c8 * 8);
  }
  wait_async0();
  __syncthreads();

  v8f acc[4] = {};
  for (int kc = 0; kc < D_MODEL / 64; ++kc) {
    const int cur = kc & 1, nxt = cur ^ 1;
    if (kc + 1 < D_MODEL / 64) {
      const int ko = (kc + 1) * 64;
      #pragma unroll
      for (int i = 0; i < 4; ++i) {
        int lin = tid + i * 256, row = lin >> 3, c8 = lin & 7;
        async_copy_b128(lds_off(&As[nxt][row][c8 * 8]),
                        Xn + (size_t)(m0 + row) * D_MODEL + ko + c8 * 8);
      }
      #pragma unroll
      for (int i = 0; i < 2; ++i) {
        int lin = tid + i * 256, row = lin >> 3, c8 = lin & 7;
        async_copy_b128(lds_off(&Bs[nxt][row][c8 * 8]),
                        Wt + (size_t)(n0 + row) * D_MODEL + ko + c8 * 8);
      }
    }
    #pragma unroll
    for (int ksub = 0; ksub < 2; ++ksub) {
      AFrag a;
      a.h[0] = *(const v8h*)&As[cur][w * 16 + lrow][ksub * 32 + half * 8];
      a.h[1] = *(const v8h*)&As[cur][w * 16 + lrow][ksub * 32 + 16 + half * 8];
      #pragma unroll
      for (int sN = 0; sN < 4; ++sN) {
        AFrag b;
        b.h[0] = *(const v8h*)&Bs[cur][sN * 16 + lrow][ksub * 32 + half * 8];
        b.h[1] = *(const v8h*)&Bs[cur][sN * 16 + lrow][ksub * 32 + 16 + half * 8];
        acc[sN] = wmma_f16(a, b, acc[sN]);
      }
    }
    wait_async0();
    __syncthreads();
  }
  // epilogue: route to Q / K (row-major f16) / V (transposed f16)
  #pragma unroll
  for (int sN = 0; sN < 4; ++sN) {
    int col = n0 + sN * 16 + lrow;
    float bias = (col < 512) ? bq[col] : bkv[col - 512];
    #pragma unroll
    for (int r = 0; r < 8; ++r) {
      int row = m0 + w * 16 + r + 8 * half;
      float val = acc[sN][r] + bias;
      if (col < 512) {
        Qb[(size_t)row * 512 + col] = (_Float16)val;
      } else if (col < 1024) {
        Kb[(size_t)row * 512 + (col - 512)] = (_Float16)val;
      } else {
        int cv = col - 1024;
        int h = cv >> 6, d = cv & 63;
        int nb = row >> 11, sp = row & (SEQ - 1);
        Vt[((size_t)((nb * NHEADS + h) * 64 + d)) * SEQ + sp] = (_Float16)val;
      }
    }
  }
}

// ---------------- fused flash attention: 1 wave per (n,h, 16-query tile) ----------------
__launch_bounds__(128)
__global__ void attn_kernel(const _Float16* __restrict__ Qb, const _Float16* __restrict__ Kb,
                            const _Float16* __restrict__ Vt, const int* __restrict__ lens,
                            _Float16* __restrict__ Ob) {
  __shared__ __align__(16) _Float16 Ptile[4][16][32];
  const int lane = threadIdx.x & 31, w = threadIdx.x >> 5;
  const int half = lane >> 4, lrow = lane & 15;
  const int gw = blockIdx.x * 4 + w;
  const int qt = gw & 127;
  const int h  = (gw >> 7) & 7;
  const int nb = gw >> 10;
  const int len = lens[nb];

  const _Float16* qp = Qb + (size_t)(nb * SEQ) * 512 + h * 64;
  const _Float16* kp = Kb + (size_t)(nb * SEQ) * 512 + h * 64;
  const _Float16* vp = Vt + (size_t)((nb * NHEADS + h) * 64) * SEQ;

  // Q tile as two A fragments (features 0..31 and 32..63)
  const _Float16* qr = qp + (size_t)(qt * 16 + lrow) * 512;
  AFrag qa0, qa1;
  qa0.h[0] = *(const v8h*)(qr + half * 8);
  qa0.h[1] = *(const v8h*)(qr + 16 + half * 8);
  qa1.h[0] = *(const v8h*)(qr + 32 + half * 8);
  qa1.h[1] = *(const v8h*)(qr + 48 + half * 8);

  v8f o[4] = {};
  float mrow[8], lrowsum[8];
  #pragma unroll
  for (int r = 0; r < 8; ++r) { mrow[r] = NEG_INF; lrowsum[r] = 0.f; }

  int kmax = qt * 16 + 16; if (len < kmax) kmax = len;
  const int nch = (kmax + 31) >> 5;
  const float SCALE = 0.125f;  // 1/sqrt(64)

  for (int ch = 0; ch < nch; ++ch) {
    const int kb = ch * 32;
    // ---- scores: two 16-key subtiles, K-dim 64 in two 32-chunks ----
    v8f s[2] = {};
    #pragma unroll
    for (int sub = 0; sub < 2; ++sub) {
      const _Float16* krp = kp + (size_t)(kb + sub * 16 + lrow) * 512;
      AFrag b0, b1;
      b0.h[0] = *(const v8h*)(krp + half * 8);
      b0.h[1] = *(const v8h*)(krp + 16 + half * 8);
      b1.h[0] = *(const v8h*)(krp + 32 + half * 8);
      b1.h[1] = *(const v8h*)(krp + 48 + half * 8);
      s[sub] = wmma_f16(qa0, b0, s[sub]);
      s[sub] = wmma_f16(qa1, b1, s[sub]);
    }
    // ---- mask + online softmax ----
    const int key0 = kb + lrow, key1 = key0 + 16;
    #pragma unroll
    for (int r = 0; r < 8; ++r) {
      const int qg = qt * 16 + r + 8 * half;
      float v0 = s[0][r] * SCALE, v1 = s[1][r] * SCALE;
      if (key0 > qg || key0 >= len) v0 = NEG_INF;
      if (key1 > qg || key1 >= len) v1 = NEG_INF;
      float rm = redmax16(fmaxf(v0, v1));
      float mnew = fmaxf(mrow[r], rm);
      float corr = __expf(mrow[r] - mnew);
      float p0 = __expf(v0 - mnew);
      float p1 = __expf(v1 - mnew);
      lrowsum[r] = lrowsum[r] * corr + redsum16(p0 + p1);
      mrow[r] = mnew;
      #pragma unroll
      for (int dv = 0; dv < 4; ++dv) o[dv][r] *= corr;
      Ptile[w][r + 8 * half][lrow]      = (_Float16)p0;
      Ptile[w][r + 8 * half][16 + lrow] = (_Float16)p1;
    }
    // ---- P(16x32) @ V(32x64): re-fragment P from per-wave LDS tile ----
    AFrag pa;
    pa.h[0] = *(const v8h*)&Ptile[w][lrow][half * 8];
    pa.h[1] = *(const v8h*)&Ptile[w][lrow][16 + half * 8];
    #pragma unroll
    for (int dv = 0; dv < 4; ++dv) {
      const _Float16* vrp = vp + (size_t)(dv * 16 + lrow) * SEQ + kb + half * 8;
      AFrag vb;
      vb.h[0] = *(const v8h*)(vrp);
      vb.h[1] = *(const v8h*)(vrp + 16);
      o[dv] = wmma_f16(pa, vb, o[dv]);
    }
  }
  // ---- normalize + store f16 ----
  #pragma unroll
  for (int r = 0; r < 8; ++r) {
    float inv = 1.0f / lrowsum[r];
    size_t row = (size_t)(nb * SEQ + qt * 16 + r + 8 * half) * 512 + h * 64;
    #pragma unroll
    for (int dv = 0; dv < 4; ++dv)
      Ob[row + dv * 16 + lrow] = (_Float16)(o[dv][r] * inv);
  }
}

// ---------------- output projection: Attn[8192][512]f16 @ WoT[512][512]f16 + bo + residual ----------------
__launch_bounds__(256)
__global__ void gemm_out_kernel(const _Float16* __restrict__ A, const _Float16* __restrict__ Wt,
                                const float* __restrict__ bo, const float* __restrict__ xres,
                                float* __restrict__ out) {
  __shared__ __align__(16) _Float16 As[2][128][64];
  __shared__ __align__(16) _Float16 Bs[2][64][64];
  const int m0 = blockIdx.x * 128;
  const int n0 = blockIdx.y * 64;
  const int tid = threadIdx.x;
  const int lane = tid & 31, w = tid >> 5;
  const int half = lane >> 4, lrow = lane & 15;

  #pragma unroll
  for (int i = 0; i < 4; ++i) {
    int lin = tid + i * 256, row = lin >> 3, c8 = lin & 7;
    async_copy_b128(lds_off(&As[0][row][c8 * 8]),
                    A + (size_t)(m0 + row) * D_MODEL + c8 * 8);
  }
  #pragma unroll
  for (int i = 0; i < 2; ++i) {
    int lin = tid + i * 256, row = lin >> 3, c8 = lin & 7;
    async_copy_b128(lds_off(&Bs[0][row][c8 * 8]),
                    Wt + (size_t)(n0 + row) * D_MODEL + c8 * 8);
  }
  wait_async0();
  __syncthreads();

  v8f acc[4] = {};
  for (int kc = 0; kc < D_MODEL / 64; ++kc) {
    const int cur = kc & 1, nxt = cur ^ 1;
    if (kc + 1 < D_MODEL / 64) {
      const int ko = (kc + 1) * 64;
      #pragma unroll
      for (int i = 0; i < 4; ++i) {
        int lin = tid + i * 256, row = lin >> 3, c8 = lin & 7;
        async_copy_b128(lds_off(&As[nxt][row][c8 * 8]),
                        A + (size_t)(m0 + row) * D_MODEL + ko + c8 * 8);
      }
      #pragma unroll
      for (int i = 0; i < 2; ++i) {
        int lin = tid + i * 256, row = lin >> 3, c8 = lin & 7;
        async_copy_b128(lds_off(&Bs[nxt][row][c8 * 8]),
                        Wt + (size_t)(n0 + row) * D_MODEL + ko + c8 * 8);
      }
    }
    #pragma unroll
    for (int ksub = 0; ksub < 2; ++ksub) {
      AFrag a;
      a.h[0] = *(const v8h*)&As[cur][w * 16 + lrow][ksub * 32 + half * 8];
      a.h[1] = *(const v8h*)&As[cur][w * 16 + lrow][ksub * 32 + 16 + half * 8];
      #pragma unroll
      for (int sN = 0; sN < 4; ++sN) {
        AFrag b;
        b.h[0] = *(const v8h*)&Bs[cur][sN * 16 + lrow][ksub * 32 + half * 8];
        b.h[1] = *(const v8h*)&Bs[cur][sN * 16 + lrow][ksub * 32 + 16 + half * 8];
        acc[sN] = wmma_f16(a, b, acc[sN]);
      }
    }
    wait_async0();
    __syncthreads();
  }
  #pragma unroll
  for (int sN = 0; sN < 4; ++sN) {
    int col = n0 + sN * 16 + lrow;
    float bias = bo[col];
    #pragma unroll
    for (int r = 0; r < 8; ++r) {
      size_t idx = (size_t)(m0 + w * 16 + r + 8 * half) * D_MODEL + col;
      out[idx] = acc[sN][r] + bias + xres[idx];
    }
  }
}

extern "C" void kernel_launch(void* const* d_in, const int* in_sizes, int n_in,
                              void* d_out, int out_size, void* d_ws, size_t ws_size,
                              hipStream_t stream) {
  (void)in_sizes; (void)n_in; (void)out_size; (void)ws_size;
  const float* x     = (const float*)d_in[0];
  const int*   lens  = (const int*)  d_in[1];
  const float* Wq    = (const float*)d_in[2];
  const float* bq    = (const float*)d_in[3];
  const float* Wkv   = (const float*)d_in[4];
  const float* bkv   = (const float*)d_in[5];
  const float* Wo    = (const float*)d_in[6];
  const float* bo    = (const float*)d_in[7];
  const float* gamma = (const float*)d_in[8];
  const float* beta  = (const float*)d_in[9];
  float* out = (float*)d_out;

  char* ws = (char*)d_ws;
  size_t off = 0;
  auto alloc = [&](size_t bytes) -> void* {
    void* p = ws + off;
    off += (bytes + 255) & ~(size_t)255;
    return p;
  };
  _Float16* xn     = (_Float16*)alloc((size_t)MTOT * 512 * 2);
  _Float16* Qb     = (_Float16*)alloc((size_t)MTOT * 512 * 2);
  _Float16* Kb     = (_Float16*)alloc((size_t)MTOT * 512 * 2);
  _Float16* Vt     = (_Float16*)alloc((size_t)MTOT * 512 * 2);
  _Float16* attn   = (_Float16*)alloc((size_t)MTOT * 512 * 2);
  _Float16* WqkvT  = (_Float16*)alloc((size_t)1536 * 512 * 2);
  _Float16* WoT    = (_Float16*)alloc((size_t)512 * 512 * 2);

  // weight transposes to f16 N-major (B-operand layout)
  cvt_w_kernel<<<(512 * 512 + 255) / 256, 256, 0, stream>>>(Wq, WqkvT, 512, 512);
  cvt_w_kernel<<<(512 * 1024 + 255) / 256, 256, 0, stream>>>(Wkv, WqkvT + (size_t)512 * 512, 512, 1024);
  cvt_w_kernel<<<(512 * 512 + 255) / 256, 256, 0, stream>>>(Wo, WoT, 512, 512);

  // pre-LayerNorm -> f16
  ln_kernel<<<MTOT / 4, 128, 0, stream>>>(x, gamma, beta, xn);

  // QKV projection (writes Q,K row-major; V transposed)
  gemm_qkv_kernel<<<dim3(MTOT / 128, 1536 / 64), 256, 0, stream>>>(xn, WqkvT, bq, bkv, Qb, Kb, Vt);

  // fused masked flash attention
  attn_kernel<<<(NBATCH * NHEADS * (SEQ / 16)) / 4, 128, 0, stream>>>(Qb, Kb, Vt, lens, attn);

  // output projection + bias + residual (f32 out)
  gemm_out_kernel<<<dim3(MTOT / 128, 512 / 64), 256, 0, stream>>>(attn, WoT, bo, x, out);
}
